// GNNLayer_455266533861
// MI455X (gfx1250) — compile-verified, matrix operand order checked by reading, not run on hardware
//
#include <hip/hip_runtime.h>
#include <hip/hip_bf16.h>

typedef __bf16 bf16_t;
typedef __attribute__((ext_vector_type(16))) __bf16 v16bf;
typedef __attribute__((ext_vector_type(8)))  __bf16 v8bf;
typedef __attribute__((ext_vector_type(8)))  float  v8f;

#define N_NODES 4096
#define C_DIM   128
#define HEADS   4
#define D_HEAD  32
#define TILES_N (N_NODES / 16)   // 256 column tiles of the score matrix
#define SCP     4104             // padded LDS score-row stride (bf16 elems), 16B aligned
#define AWAVES  8                // waves per attention workgroup

// ---- LDS layout for the attention kernel (dynamic shared) ----
#define SC_BYTES     (16 * SCP * 2)          // 131328: bf16 scores [16][SCP]
#define OFF_RED      (SC_BYTES)              // float[256] partial reductions (1024B)
#define OFF_RMAX     (OFF_RED + 1024)        // float[16]
#define OFF_RINV     (OFF_RMAX + 64)         // float[16]
#define OFF_POUT     (OFF_RINV + 64)         // float[8*2*32*8] = 16384B
#define SMEM_BYTES   (OFF_POUT + 16384)      // 148864 total -> 2 WGs per 320KB WGP

// Load a 16x32 bf16 A-fragment (or 32x16 B-fragment, same lane packing) from a
// row-major source: lane r (0-15) maps to row r; lanes 0-15 carry K 0..7 &
// 16..23, lanes 16-31 carry K 8..15 & 24..31 (ISA 7.12.2). Two 128-bit loads.
static __device__ __forceinline__ v16bf load_frag(const bf16_t* __restrict__ base, int stride) {
  const int lane = threadIdx.x & 31;
  const int r    = lane & 15;
  const int half = lane >> 4;
  const bf16_t* p = base + r * stride + half * 8;
  v8bf lo = *(const v8bf*)(p);
  v8bf hi = *(const v8bf*)(p + 16);
  v16bf f;
#pragma unroll
  for (int i = 0; i < 8; ++i) { f[i] = lo[i]; f[i + 8] = hi[i]; }
  return f;
}

static __device__ __forceinline__ v8f wmma_bf16(v16bf a, v16bf b, v8f c) {
  return __builtin_amdgcn_wmma_f32_16x16x32_bf16(false, a, false, b, (short)0, c, false, false);
}

// ---------------- mask bitmap ----------------
__global__ void k_zero(unsigned* __restrict__ p, int n) {
  for (int i = blockIdx.x * blockDim.x + threadIdx.x; i < n; i += gridDim.x * blockDim.x)
    p[i] = 0u;
}

// bit index = key*4096 + query  (key-major so one word covers 32 query rows)
__global__ void k_build_mask(const int* __restrict__ ei, unsigned* __restrict__ bm, int E) {
  int e = blockIdx.x * blockDim.x + threadIdx.x;
  if (e >= E) return;
  int r = ei[e];       // query node
  int c = ei[E + e];   // key node
  atomicOr(&bm[(c << 7) | (r >> 5)], 1u << (r & 31));
}

__global__ void k_f2bf(const float* __restrict__ s, bf16_t* __restrict__ d, int n) {
  for (int i = blockIdx.x * blockDim.x + threadIdx.x; i < n; i += gridDim.x * blockDim.x)
    d[i] = (bf16_t)s[i];
}

// ---------------- QKV projections ----------------
// out[n,c] = sum_k x[n,k] * w[c,k] + b[c]; q pre-scaled by 1/sqrt(D); v stored transposed [C,N]
__global__ void k_qkv(const bf16_t* __restrict__ xbf, const bf16_t* __restrict__ wbf,
                      const float* __restrict__ bq, const float* __restrict__ bk,
                      const float* __restrict__ bv,
                      bf16_t* __restrict__ q, bf16_t* __restrict__ k, bf16_t* __restrict__ vT) {
  const int wave = threadIdx.x >> 5, lane = threadIdx.x & 31;
  const int half = lane >> 4, c16 = lane & 15;
  const int tile = blockIdx.x * 4 + wave;          // 0..2047
  const int z = blockIdx.y;                        // 0=q 1=k 2=v
  const int rowBase = (tile >> 3) * 16;
  const int colBase = (tile & 7) * 16;
  const bf16_t* w = wbf + z * (C_DIM * C_DIM);
  const float* bias = (z == 0) ? bq : (z == 1) ? bk : bv;

  v8f acc = {};
#pragma unroll
  for (int kk = 0; kk < 4; ++kk) {
    v16bf a = load_frag(xbf + rowBase * C_DIM + kk * 32, C_DIM);
    v16bf b = load_frag(w   + colBase * C_DIM + kk * 32, C_DIM);
    acc = wmma_bf16(a, b, acc);
  }
  const int col = colBase + c16;
  const float bz = bias[col];
  if (z == 2) {                       // V transposed: 8 consecutive nodes -> one 16B store
    v8bf o;
#pragma unroll
    for (int i = 0; i < 8; ++i) o[i] = (bf16_t)(acc[i] + bz);
    *(v8bf*)(vT + col * N_NODES + rowBase + half * 8) = o;
  } else {
    bf16_t* dst = (z == 0) ? q : k;
    const float scl = (z == 0) ? 0.17677669529663687f : 1.0f;  // 1/sqrt(32)
#pragma unroll
    for (int i = 0; i < 8; ++i) {
      int row = rowBase + i + half * 8;
      dst[row * C_DIM + col] = (bf16_t)((acc[i] + bz) * scl);
    }
  }
}

// ---------------- masked attention (scores + softmax + PV) ----------------
// One workgroup (8 waves) per (16-row query tile, head). Full 16x4096 bf16
// score stripe lives in LDS (CDNA5: 320KB/WGP makes this possible).
__global__ void k_attn(const bf16_t* __restrict__ q, const bf16_t* __restrict__ k,
                       const bf16_t* __restrict__ vT, const unsigned* __restrict__ bm,
                       bf16_t* __restrict__ attnOut) {
  extern __shared__ char smem[];
  bf16_t* sc     = (bf16_t*)smem;
  float*  redf   = (float*)(smem + OFF_RED);
  float*  rowMax = (float*)(smem + OFF_RMAX);
  float*  rowInv = (float*)(smem + OFF_RINV);
  float*  pOut   = (float*)(smem + OFF_POUT);

  const int rowBase = blockIdx.x * 16;
  const int h = blockIdx.y;
  const int wave = threadIdx.x >> 5, lane = threadIdx.x & 31;
  const int half = lane >> 4, c16 = lane & 15;

  // ---- Phase 1: S = Q K^T tiles via WMMA, mask, park bf16 scores in LDS ----
  v16bf qa = load_frag(q + rowBase * C_DIM + h * D_HEAD, C_DIM);
  const int wordCol = rowBase >> 5;
  const int shift   = rowBase & 31;
  for (int t = wave; t < TILES_N; t += AWAVES) {
    if (t + AWAVES < TILES_N)   // pull next K tile toward the WGP cache
      __builtin_prefetch(k + ((t + AWAVES) * 16 + c16) * C_DIM + h * D_HEAD + half * 8, 0, 3);
    int kcol = t * 16 + c16;
    v16bf kb = load_frag(k + (t * 16) * C_DIM + h * D_HEAD, C_DIM);
    v8f s = {};
    s = wmma_bf16(qa, kb, s);
    unsigned bits = bm[(kcol << 7) | wordCol] >> shift;   // 16 query-row bits
#pragma unroll
    for (int i = 0; i < 8; ++i) {
      int tt = i + half * 8;                               // query row in tile
      float sv = ((bits >> tt) & 1u) ? s[i] : -__builtin_inff();
      sc[tt * SCP + kcol] = (bf16_t)sv;
    }
  }
  __syncthreads();

  // ---- Phase 2: per-row max and 1/sum(exp), vectorized b128 LDS reads ----
  {
    int row = threadIdx.x >> 4;            // 16 rows
    int sub = threadIdx.x & 15;            // 16 segments of 256 cols
    const v8bf* rp = (const v8bf*)(sc + row * SCP + sub * 256);
    float m = -__builtin_inff();
#pragma unroll 4
    for (int c = 0; c < 32; ++c) {
      v8bf v = rp[c];
#pragma unroll
      for (int j = 0; j < 8; ++j) m = fmaxf(m, (float)v[j]);
    }
    redf[row * 16 + sub] = m;
    __syncthreads();
    if (sub == 0) {
      float mm = -__builtin_inff();
      for (int j = 0; j < 16; ++j) mm = fmaxf(mm, redf[row * 16 + j]);
      rowMax[row] = mm;
    }
    __syncthreads();
    float rmax = rowMax[row];
    float sum = 0.f;
#pragma unroll 4
    for (int c = 0; c < 32; ++c) {
      v8bf v = rp[c];
#pragma unroll
      for (int j = 0; j < 8; ++j) sum += __expf((float)v[j] - rmax);
    }
    redf[row * 16 + sub] = sum;
    __syncthreads();
    if (sub == 0) {
      float ss = 0.f;
      for (int j = 0; j < 16; ++j) ss += redf[row * 16 + j];
      rowInv[row] = 1.0f / ss;
    }
    __syncthreads();
  }

  // ---- Phase 3: O = exp(S - max) @ V via WMMA (exp co-executes as TRANS) ----
  v8f O0 = {}, O1 = {};
  {
    float rmax = rowMax[c16];   // A-fragment: lane's row is (lane&15) for all K
    for (int ch = wave; ch < 128; ch += AWAVES) {
      int mBase = ch * 32;
      if (ch + AWAVES < 128) {  // prefetch next V chunk (two d-halves)
        __builtin_prefetch(vT + (h * D_HEAD + c16)      * N_NODES + mBase + AWAVES * 32, 0, 3);
        __builtin_prefetch(vT + (h * D_HEAD + 16 + c16) * N_NODES + mBase + AWAVES * 32, 0, 3);
      }
      v16bf a = load_frag(sc + mBase, SCP);
      v16bf pa;
#pragma unroll
      for (int i = 0; i < 16; ++i) pa[i] = (bf16_t)__expf((float)a[i] - rmax);
      v16bf b0 = load_frag(vT + (h * D_HEAD)      * N_NODES + mBase, N_NODES);
      v16bf b1 = load_frag(vT + (h * D_HEAD + 16) * N_NODES + mBase, N_NODES);
      O0 = wmma_bf16(pa, b0, O0);
      O1 = wmma_bf16(pa, b1, O1);
    }
  }
  // ---- Phase 4: cross-wave reduce (waves 0/1 take one output tile each) ----
#pragma unroll
  for (int i = 0; i < 8; ++i) {
    pOut[((wave * 2 + 0) * 32 + lane) * 8 + i] = O0[i];
    pOut[((wave * 2 + 1) * 32 + lane) * 8 + i] = O1[i];
  }
  __syncthreads();
  if (wave < 2) {
    const int t = wave;
#pragma unroll
    for (int i = 0; i < 8; ++i) {
      float v = 0.f;
#pragma unroll
      for (int w = 0; w < AWAVES; ++w) v += pOut[((w * 2 + t) * 32 + lane) * 8 + i];
      int row = i + half * 8;
      attnOut[(rowBase + row) * C_DIM + h * D_HEAD + t * 16 + c16] =
          (bf16_t)(v * rowInv[row]);
    }
  }
}

// ---------------- output projection + residual + LayerNorm ----------------
__global__ void k_outln(const bf16_t* __restrict__ attnbf, const bf16_t* __restrict__ wo_bf,
                        const float* __restrict__ bo, const float* __restrict__ x,
                        const float* __restrict__ gamma, const float* __restrict__ beta,
                        float* __restrict__ out) {
  __shared__ float yb[16 * C_DIM];
  __shared__ float redS[256], redQ[256];
  __shared__ float mu_s[16], rs_s[16];
  const int rowBase = blockIdx.x * 16;
  const int wave = threadIdx.x >> 5, lane = threadIdx.x & 31;
  const int half = lane >> 4, c16 = lane & 15;
  const int colBase = wave * 16;

  v8f acc = {};
#pragma unroll
  for (int kk = 0; kk < 4; ++kk) {
    v16bf a = load_frag(attnbf + rowBase * C_DIM + kk * 32, C_DIM);
    v16bf b = load_frag(wo_bf  + colBase * C_DIM + kk * 32, C_DIM);
    acc = wmma_bf16(a, b, acc);
  }
  const int col = colBase + c16;
  const float bz = bo[col];
#pragma unroll
  for (int i = 0; i < 8; ++i) {
    int row = i + half * 8;
    yb[row * C_DIM + col] = acc[i] + bz + x[(rowBase + row) * C_DIM + col];
  }
  __syncthreads();

  const int row = threadIdx.x >> 4, sub = threadIdx.x & 15;
  float s = 0.f, sq = 0.f;
#pragma unroll
  for (int j = 0; j < 8; ++j) {
    float v = yb[row * C_DIM + sub * 8 + j];
    s += v; sq += v * v;
  }
  redS[threadIdx.x] = s; redQ[threadIdx.x] = sq;
  __syncthreads();
  if (sub == 0) {
    float ts = 0.f, tq = 0.f;
    for (int j = 0; j < 16; ++j) { ts += redS[row * 16 + j]; tq += redQ[row * 16 + j]; }
    float mu = ts / (float)C_DIM;
    float var = tq / (float)C_DIM - mu * mu;
    mu_s[row] = mu;
    rs_s[row] = rsqrtf(var + 1e-5f);
  }
  __syncthreads();
  const float mu = mu_s[row], rs = rs_s[row];
#pragma unroll
  for (int j = 0; j < 8; ++j) {
    int c = sub * 8 + j;
    float v = yb[row * C_DIM + c];
    out[(rowBase + row) * C_DIM + c] = (v - mu) * rs * gamma[c] + beta[c];
  }
}

extern "C" void kernel_launch(void* const* d_in, const int* in_sizes, int n_in,
                              void* d_out, int out_size, void* d_ws, size_t ws_size,
                              hipStream_t stream) {
  const float* x     = (const float*)d_in[0];
  const int*   ei    = (const int*)d_in[1];
  const float* wq    = (const float*)d_in[2];
  const float* bq    = (const float*)d_in[3];
  const float* wk    = (const float*)d_in[4];
  const float* bk    = (const float*)d_in[5];
  const float* wv    = (const float*)d_in[6];
  const float* bv    = (const float*)d_in[7];
  const float* wo    = (const float*)d_in[8];
  const float* bo    = (const float*)d_in[9];
  const float* gamma = (const float*)d_in[10];
  const float* beta  = (const float*)d_in[11];
  float* out = (float*)d_out;
  const int E = in_sizes[1] / 2;

  char* ws = (char*)d_ws;
  unsigned* maskBits = (unsigned*)(ws);                    // 2 MB bitmap
  bf16_t* xbf = (bf16_t*)(ws + (2u << 20));                // 1 MB
  bf16_t* qbf = (bf16_t*)(ws + (3u << 20));                // 1 MB
  bf16_t* kbf = (bf16_t*)(ws + (4u << 20));                // 1 MB
  bf16_t* vT  = (bf16_t*)(ws + (5u << 20));                // 1 MB  (transposed V [C,N])
  bf16_t* abf = (bf16_t*)(ws + (6u << 20));                // 1 MB  (attention output, bf16)
  bf16_t* wbf = (bf16_t*)(ws + (7u << 20));                // 4 * 32KB bf16 weights

  hipFuncSetAttribute((const void*)k_attn,
                      hipFuncAttributeMaxDynamicSharedMemorySize, SMEM_BYTES);

  const int words = (N_NODES * N_NODES) / 32;
  k_zero<<<512, 256, 0, stream>>>(maskBits, words);
  k_build_mask<<<(E + 255) / 256, 256, 0, stream>>>(ei, maskBits, E);
  k_f2bf<<<512, 256, 0, stream>>>(x, xbf, N_NODES * C_DIM);
  k_f2bf<<<64, 256, 0, stream>>>(wq, wbf + 0 * C_DIM * C_DIM, C_DIM * C_DIM);
  k_f2bf<<<64, 256, 0, stream>>>(wk, wbf + 1 * C_DIM * C_DIM, C_DIM * C_DIM);
  k_f2bf<<<64, 256, 0, stream>>>(wv, wbf + 2 * C_DIM * C_DIM, C_DIM * C_DIM);
  k_f2bf<<<64, 256, 0, stream>>>(wo, wbf + 3 * C_DIM * C_DIM, C_DIM * C_DIM);

  k_qkv<<<dim3(512, 3), 128, 0, stream>>>(xbf, wbf, bq, bk, bv, qbf, kbf, vT);
  k_attn<<<dim3(TILES_N, HEADS), 32 * AWAVES, SMEM_BYTES, stream>>>(qbf, kbf, vT, maskBits, abf);
  k_outln<<<TILES_N, 256, 0, stream>>>(abf, wbf + 3 * C_DIM * C_DIM, bo, x, gamma, beta, out);
}
